// AttentionLayer_sigmoid_62534723830199
// MI455X (gfx1250) — compile-verified, matrix operand order checked by reading, not run on hardware
//
#include <hip/hip_runtime.h>
#include <math.h>

// ---- problem constants (match reference) ----
#define IN_DIM 768
#define HID    768
#define BATCH  32
#define SEQ    2048
#define MTILE  32              // memory_bank rows per workgroup
#define KSTEPS (IN_DIM / 32)   // 24 K-steps of 32 for bf16 WMMA
#define NTILES 3               // 3 tiles of 256 columns (8 waves: 2 M x 4 N-groups of 64)

typedef __attribute__((ext_vector_type(16))) __bf16 v16bf;
typedef __attribute__((ext_vector_type(8)))  float  v8f;

static __device__ __forceinline__ unsigned short f2bf(float f) {
    // round-to-nearest-even f32 -> bf16
    unsigned int x = __float_as_uint(f);
    unsigned int r = x + 0x7FFFu + ((x >> 16) & 1u);
    return (unsigned short)(r >> 16);
}

// ---------------------------------------------------------------------------
// Kernel 0: zero output (we accumulate into it with global atomics)
// ---------------------------------------------------------------------------
__global__ void zero_out_kernel(float* __restrict__ out) {
    int i = blockIdx.x * 256 + threadIdx.x;
    if (i < BATCH * IN_DIM) out[i] = 0.0f;
}

// ---------------------------------------------------------------------------
// Kernel 1: pack W1 [H,D] f32 -> bf16 in WMMA B-fragment order.
// Fragment order: idx = ((n16*KSTEPS + k32)*32 + lane)*16 + j
//   lane -> (hi = lane>=16, n = lane&15), half j -> kk per ISA 16-bit layout:
//   kk = (j&7) + hi*8 + (j>=8 ? 16 : 0)
// so each lane's 16 halves are 32 contiguous bytes -> global_load_b128 x2.
// ---------------------------------------------------------------------------
__global__ __launch_bounds__(256) void pack_w1_kernel(const float* __restrict__ W1,
                                                      unsigned short* __restrict__ w1p) {
    int idx  = blockIdx.x * 256 + threadIdx.x;      // 0 .. 768*768-1
    int j    = idx & 15;
    int lane = (idx >> 4) & 31;
    int k32  = (idx >> 9) % KSTEPS;
    int n16  = (idx >> 9) / KSTEPS;
    int hi   = lane >> 4;
    int n    = lane & 15;
    int kk   = (j & 7) + hi * 8 + ((j >= 8) ? 16 : 0);
    int h    = n16 * 16 + n;
    int d    = k32 * 32 + kk;
    w1p[idx] = f2bf(W1[(size_t)h * IN_DIM + d]);
}

// ---------------------------------------------------------------------------
// Kernel 2: uh[b,h] = input[b,:] . W2[h,:] + b2[h]   (tiny, 38 MFLOP)
// ---------------------------------------------------------------------------
__global__ __launch_bounds__(256) void uh_kernel(const float* __restrict__ input,
                                                 const float* __restrict__ W2,
                                                 const float* __restrict__ b2,
                                                 float* __restrict__ uh) {
    __shared__ float inrow[IN_DIM];
    int b = blockIdx.x;
    for (int d = threadIdx.x; d < IN_DIM; d += 256)
        inrow[d] = input[b * IN_DIM + d];
    __syncthreads();
    for (int h = threadIdx.x; h < HID; h += 256) {
        const float4* wrow = (const float4*)(W2 + (size_t)h * IN_DIM);
        const float4* irow = (const float4*)inrow;
        float acc = b2[h];
        #pragma unroll 4
        for (int d4 = 0; d4 < IN_DIM / 4; ++d4) {
            float4 wv = wrow[d4];
            float4 iv = irow[d4];
            acc += wv.x * iv.x + wv.y * iv.y + wv.z * iv.z + wv.w * iv.w;
        }
        uh[b * HID + h] = acc;
    }
}

// ---------------------------------------------------------------------------
// Kernel 3: fused attention. One WG per (b, 32-row tile of L).
//   - convert tile f32->bf16 into LDS in A-fragment order
//   - WMMA GEMM (bf16, f32 acc): 8 waves = 2 (M16) x 4 (64-col groups), 3 N-tiles
//   - epilogue: tanh(c+uh)*v, ds_add_f32 row reduction, sigmoid*mask
//   - pooling: out[b,d] += sum_l align[l]*mem[b,l,d] via global atomic f32 add
// ---------------------------------------------------------------------------
__global__ __launch_bounds__(256) void attn_main_kernel(const float* __restrict__ mem,
                                                        const int*   __restrict__ mask,
                                                        const float* __restrict__ uh,
                                                        const float* __restrict__ vvec,
                                                        const unsigned short* __restrict__ w1p,
                                                        float* __restrict__ out) {
    __shared__ alignas(32) unsigned short A_lds[2 * KSTEPS * 512]; // 48 KB
    __shared__ float zrow[MTILE];
    __shared__ float alig[MTILE];

    const int tid = threadIdx.x;
    const int b   = blockIdx.x >> 6;        // 64 row-tiles per batch
    const int lt  = blockIdx.x & 63;
    const int l0  = lt * MTILE;

    if (tid < MTILE) zrow[tid] = 0.0f;

    // ---- stage A tile into LDS, bf16, A-fragment order ----
    const float* srcbase = mem + ((size_t)b * SEQ + l0) * IN_DIM;
    #pragma unroll 4
    for (int i = 0; i < (MTILE * IN_DIM / 2) / 256; ++i) {
        int p  = tid + i * 256;             // pair index
        int ll = p / (IN_DIM / 2);          // local row 0..31
        int d  = (p % (IN_DIM / 2)) * 2;    // even column
        float2 f = *(const float2*)(srcbase + (size_t)ll * IN_DIM + d);
        unsigned int packed = (unsigned int)f2bf(f.x) | ((unsigned int)f2bf(f.y) << 16);
        int m16 = ll >> 4, row = ll & 15;
        int k32 = d >> 5,  kk  = d & 31;
        int hi  = (kk >> 3) & 1;                         // 0,1,0,1 per octet
        int j   = (kk & 7) + ((kk >= 16) ? 8 : 0);
        int lane = row + hi * 16;
        int haddr = ((m16 * KSTEPS + k32) << 9) + lane * 16 + j; // j even
        *(unsigned int*)&A_lds[haddr] = packed;
    }
    __syncthreads();

    // ---- WMMA GEMM + epilogue ----
    const int w    = tid >> 5;
    const int lane = tid & 31;
    const int m16  = w & 1;        // which 16 rows
    const int ngrp = w >> 1;       // which 64-col group within an N-tile
    const int hi8  = (lane >> 4) * 8;
    const int ncol = lane & 15;

    float zacc[8];
    #pragma unroll
    for (int r = 0; r < 8; ++r) zacc[r] = 0.0f;

    for (int nt = 0; nt < NTILES; ++nt) {
        v8f acc[4];
        #pragma unroll
        for (int a = 0; a < 4; ++a)
            #pragma unroll
            for (int r = 0; r < 8; ++r) acc[a][r] = 0.0f;

        const int hbase0 = nt * 256 + ngrp * 64;
        for (int k = 0; k < KSTEPS; ++k) {
            v16bf afrag = *(const v16bf*)&A_lds[((m16 * KSTEPS + k) << 9) + lane * 16];
            #pragma unroll
            for (int a = 0; a < 4; ++a) {
                int n16g = (hbase0 >> 4) + a;
                v16bf bfrag = *(const v16bf*)&w1p[(((size_t)n16g * KSTEPS + k) * 32 + lane) * 16];
                acc[a] = __builtin_amdgcn_wmma_f32_16x16x32_bf16(
                    /*neg_a=*/false, afrag, /*neg_b=*/false, bfrag,
                    /*c_mod=*/(short)0, acc[a], /*reuse_a=*/false, /*reuse_b=*/false);
            }
        }
        // epilogue: t = tanh(c + uh[h]); z[row] += t * v[h]
        #pragma unroll
        for (int a = 0; a < 4; ++a) {
            int h = hbase0 + a * 16 + ncol;
            float uhv = uh[b * HID + h];
            float vv  = vvec[h];
            #pragma unroll
            for (int r = 0; r < 8; ++r) {
                float t = tanhf(acc[a][r] + uhv);
                zacc[r] += t * vv;
            }
        }
    }

    // reduce per-lane row partials across lanes/waves via LDS float atomics
    #pragma unroll
    for (int r = 0; r < 8; ++r)
        atomicAdd(&zrow[m16 * 16 + hi8 + r], zacc[r]);
    __syncthreads();

    if (tid < MTILE) {
        float z = zrow[tid];
        float s = 1.0f / (1.0f + __expf(-z));
        int mk  = mask[(size_t)b * SEQ + l0 + tid];
        alig[tid] = s * (float)mk;
    }
    __syncthreads();

    // ---- gated pooling over the 32 rows (f32, L2-resident re-read) ----
    for (int d = tid; d < IN_DIM; d += 256) {
        const float* col = srcbase + d;
        float acc = 0.0f;
        #pragma unroll 8
        for (int l = 0; l < MTILE; ++l)
            acc += alig[l] * col[(size_t)l * IN_DIM];
        atomicAdd(&out[b * IN_DIM + d], acc);
    }
}

// ---------------------------------------------------------------------------
extern "C" void kernel_launch(void* const* d_in, const int* in_sizes, int n_in,
                              void* d_out, int out_size, void* d_ws, size_t ws_size,
                              hipStream_t stream) {
    const float* input = (const float*)d_in[0];
    const float* mem   = (const float*)d_in[1];
    const int*   mask  = (const int*)d_in[2];
    const float* W1    = (const float*)d_in[3];
    const float* W2    = (const float*)d_in[4];
    const float* b2    = (const float*)d_in[5];
    const float* v     = (const float*)d_in[6];
    float* out = (float*)d_out;

    // workspace: uh [32*768 f32] then packed bf16 W1 [768*768]
    float* uh = (float*)d_ws;
    unsigned short* w1p = (unsigned short*)((char*)d_ws + (size_t)BATCH * HID * sizeof(float));

    zero_out_kernel<<<(BATCH * IN_DIM + 255) / 256, 256, 0, stream>>>(out);
    pack_w1_kernel<<<(HID * IN_DIM) / 256, 256, 0, stream>>>(W1, w1p);
    uh_kernel<<<BATCH, 256, 0, stream>>>(input, W2, b2, uh);
    attn_main_kernel<<<BATCH * (SEQ / MTILE), 256, 0, stream>>>(mem, mask, uh, v, w1p, out);
}
